// PhotonicInterferenceLayer_11390253269476
// MI455X (gfx1250) — matrix operand order repaired and checked by reading, not compiled, not run on hardware
//
#include <hip/hip_runtime.h>
#include <hip/hip_bf16.h>

typedef __attribute__((ext_vector_type(16))) _Float16 v16h;
typedef __attribute__((ext_vector_type(8)))  float    v8f;

#define COUPLING 0.12f
#define EPSV 1e-8f

// ---------------------------------------------------------------------------
// Fast tanh: gfx1250 has V_TANH_F32 (trans pipe, co-executes with WMMA).
// Hardware tanh saturates for any finite input -> no clamp needed.
// Fallback: tanh(x) = (e^{2x}-1)/(e^{2x}+1) via v_exp_f32 + v_rcp_f32,
// which DOES need a clamp to avoid inf/inf -> NaN.
// ---------------------------------------------------------------------------
__device__ __forceinline__ float fast_tanh(float x) {
#if __has_builtin(__builtin_amdgcn_tanhf)
    return __builtin_amdgcn_tanhf(x);
#elif __has_builtin(__builtin_amdgcn_tanh_f32)
    return __builtin_amdgcn_tanh_f32(x);
#else
    x = fminf(fmaxf(x, -15.0f), 15.0f);
    float e = __builtin_amdgcn_exp2f(x * 2.885390081777927f); // e^(2x)
    return (e - 1.0f) * __builtin_amdgcn_rcpf(e + 1.0f);
#endif
}

// ---------------------------------------------------------------------------
// Zero the 128-float column-sum accumulator in workspace (ws is poisoned).
// ---------------------------------------------------------------------------
__global__ void pil_zero_kernel(float* __restrict__ p, int n) {
    int i = blockIdx.x * blockDim.x + threadIdx.x;
    if (i < n) p[i] = 0.0f;
}

// ---------------------------------------------------------------------------
// Main kernel: S[oo] = sum_p tanh( z_flat[p,0:32] . W2[oo] + cv[oo] )
//   W2[oo][k] = heads_w[oo*64 + 32 + k]   (the z_past half of the weights)
//   cv[oo]    = heads_b[oo] + sum_d cand[d]*heads_w[oo*64 + d]
// One wave: 8 column tiles (128 outputs), grid-stride over P/16 row tiles.
// Uses v_wmma_f32_16x16x32_f16 (D = A(16x32) * B(32x16) + 0), f32 accum.
// ---------------------------------------------------------------------------
__global__ void __launch_bounds__(256)
pil_accum_kernel(const float* __restrict__ cand,
                 const float* __restrict__ zpast,
                 const float* __restrict__ heads_w,
                 const float* __restrict__ heads_b,
                 float* __restrict__ gsum,
                 int P)
{
    const int lane   = threadIdx.x & 31;
    const int n      = lane & 15;        // output column within tile
    const int g      = lane >> 4;        // half-wave group (K-half selector)
    const int waveId = (blockIdx.x * blockDim.x + threadIdx.x) >> 5;
    const int nWaves = (gridDim.x * blockDim.x) >> 5;

    // --- Per-wave constant setup: B fragments (f16) + scalar C offsets -----
    // B fragment layout (32x16, 16-bit): lane n holds column n; lanes 0-15
    // hold K={0..7,16..23}, lanes 16-31 hold K={8..15,24..31}.
    v16h  bfrag[8];
    float cvs[8];
#pragma unroll
    for (int t = 0; t < 8; ++t) {
        const int oo = t * 16 + n;                 // combined output index
        const float* wrow = heads_w + oo * 64;
#pragma unroll
        for (int j = 0; j < 16; ++j)
            bfrag[t][j] = (_Float16)wrow[32 + 8 * g + (j & 7) + 16 * (j >> 3)];
        float cv = heads_b[oo];
#pragma unroll
        for (int d = 0; d < 32; ++d)
            cv += cand[d] * wrow[d];
        cvs[t] = cv;
    }

    v8f acc[8];
#pragma unroll
    for (int t = 0; t < 8; ++t)
#pragma unroll
        for (int j = 0; j < 8; ++j) acc[t][j] = 0.0f;

    const v8f czero = {};

    // --- Stream row tiles ---------------------------------------------------
    // A fragment (16x32 f16): lane m=lane&15 holds row m; lanes 0-15 need
    // K={0..7,16..23}, lanes 16-31 need K={8..15,24..31}; with that load
    // order the local float index equals the half index j exactly.
    const int nTiles = P >> 4;
    for (int tile = waveId; tile < nTiles; tile += nWaves) {
        const int row = tile * 16 + n;
        const float* xr = zpast + (size_t)row * 32 + g * 8;
        float4 L0 = *(const float4*)(xr + 0);
        float4 L1 = *(const float4*)(xr + 4);
        float4 L2 = *(const float4*)(xr + 16);
        float4 L3 = *(const float4*)(xr + 20);
        v16h a;
        a[0]  = (_Float16)L0.x;  a[1]  = (_Float16)L0.y;
        a[2]  = (_Float16)L0.z;  a[3]  = (_Float16)L0.w;
        a[4]  = (_Float16)L1.x;  a[5]  = (_Float16)L1.y;
        a[6]  = (_Float16)L1.z;  a[7]  = (_Float16)L1.w;
        a[8]  = (_Float16)L2.x;  a[9]  = (_Float16)L2.y;
        a[10] = (_Float16)L2.z;  a[11] = (_Float16)L2.w;
        a[12] = (_Float16)L3.x;  a[13] = (_Float16)L3.y;
        a[14] = (_Float16)L3.z;  a[15] = (_Float16)L3.w;

#pragma unroll
        for (int t = 0; t < 8; ++t) {
            v8f d = __builtin_amdgcn_wmma_f32_16x16x32_f16(
                false, a, false, bfrag[t], (short)0, czero, false, false);
#pragma unroll
            for (int j = 0; j < 8; ++j)
                acc[t][j] += fast_tanh(d[j] + cvs[t]);
        }
    }

    // --- Column-sum reduction: D element (M=j+8g, N=lane&15) ----------------
#pragma unroll
    for (int t = 0; t < 8; ++t) {
        float s = 0.0f;
#pragma unroll
        for (int j = 0; j < 8; ++j) s += acc[t][j];
        s += __shfl_xor(s, 16, 32);          // combine M-halves (lane ^ 16)
        if (g == 0)
            atomicAdd(&gsum[t * 16 + n], s);
    }
}

// ---------------------------------------------------------------------------
// Epilogue: mean, out_w matvec, analytic rel_bias mean, tanh, normalize.
// ---------------------------------------------------------------------------
__global__ void pil_epilogue_kernel(const float* __restrict__ cand,
                                    const float* __restrict__ out_w,
                                    const float* __restrict__ out_b,
                                    const float* __restrict__ rel_bias,
                                    const float* __restrict__ norm_scale,
                                    const int*   __restrict__ posp,
                                    const float* __restrict__ gsum,
                                    float* __restrict__ out,
                                    int P)
{
    __shared__ float mean[128];
    __shared__ float zsh[32];
    __shared__ float bias[2];
    __shared__ float maxint_sh;

    const int tid  = threadIdx.x;
    const float invP = 1.0f / (float)P;

    if (tid < 128) mean[tid] = gsum[tid] * invP;

    if (tid == 0) {
        // idx = clip((pos-P+r) + 64, 0, 128) for r in [0,P)
        long long pos = (long long)(*posp);
        long long cl = (long long)P - pos - 64 + 1;     // occurrences of idx==0
        if (cl < 0) cl = 0; if (cl > P) cl = P;
        long long ch = pos - 64;                        // occurrences of idx==128
        if (ch < 0) ch = 0; if (ch > P) ch = P;
        float b0 = (float)cl * rel_bias[0]       + (float)ch * rel_bias[128 * 2];
        float b1 = (float)cl * rel_bias[1]       + (float)ch * rel_bias[128 * 2 + 1];
        for (int k = 1; k < 128; ++k) {                 // middle: each idx==k at most once
            long long r = (long long)k - 64 + (long long)P - pos;
            if (r >= 0 && r < (long long)P) {
                b0 += rel_bias[2 * k];
                b1 += rel_bias[2 * k + 1];
            }
        }
        bias[0] = b0 * invP;
        bias[1] = b1 * invP;
    }
    __syncthreads();

    if (tid < 32) {
        float o = out_b[tid];
#pragma unroll 8
        for (int j = 0; j < 128; ++j)
            o += out_w[tid * 128 + j] * mean[j];
        zsh[tid] = fast_tanh(cand[tid] + COUPLING * (o + bias[tid & 1]));
    }
    __syncthreads();

    if (tid == 0) {
        float m = 0.0f;
        for (int d = 0; d < 16; ++d) {
            float s = zsh[2 * d] * zsh[2 * d] + zsh[2 * d + 1] * zsh[2 * d + 1];
            m = fmaxf(m, s);
        }
        maxint_sh = m;
    }
    __syncthreads();

    if (tid < 32) {
        float m = maxint_sh;
        float scale = norm_scale[0] * rsqrtf(m + EPSV);
        out[tid] = (m > 0.0f) ? zsh[tid] * scale : zsh[tid];
    }
}

// ---------------------------------------------------------------------------
extern "C" void kernel_launch(void* const* d_in, const int* in_sizes, int n_in,
                              void* d_out, int out_size, void* d_ws, size_t ws_size,
                              hipStream_t stream) {
    const float* cand       = (const float*)d_in[0];   // [16,2]
    const float* zpast      = (const float*)d_in[1];   // [P,16,2]
    const float* heads_w    = (const float*)d_in[2];   // [4,32,64]
    const float* heads_b    = (const float*)d_in[3];   // [4,32]
    const float* out_w      = (const float*)d_in[4];   // [32,128]
    const float* out_b      = (const float*)d_in[5];   // [32]
    const float* rel_bias   = (const float*)d_in[6];   // [129,2]
    const float* norm_scale = (const float*)d_in[7];   // [1]
    const int*   posp       = (const int*)d_in[8];     // [1]
    float*       out        = (float*)d_out;           // [16,2] = 32 floats
    float*       gsum       = (float*)d_ws;            // 128 floats of scratch

    const int P = in_sizes[1] / 32;                    // 524288

    pil_zero_kernel<<<1, 128, 0, stream>>>(gsum, 128);

    // 512 blocks x 8 waves = 4096 waves; P/16 = 32768 row tiles -> 8 per wave.
    pil_accum_kernel<<<512, 256, 0, stream>>>(cand, zpast, heads_w, heads_b,
                                              gsum, P);

    pil_epilogue_kernel<<<1, 128, 0, stream>>>(cand, out_w, out_b, rel_bias,
                                               norm_scale, posp, gsum, out, P);
}
